// MultiheadAttention_87187836109262
// MI455X (gfx1250) — matrix operand order repaired and checked by reading, not compiled, not run on hardware
//
#include <hip/hip_runtime.h>
#include <hip/hip_bf16.h>
#include <math.h>

// ---------------------------------------------------------------------------
// WavLM-style multihead attention for MI455X (gfx1250), wave32 + WMMA f16.
//   S=1024, B=4, E=1024, H=16, D=64, NUM_BUCKETS=32, MAX_DISTANCE=128
// Pipeline:
//   1. f32->f16 convert (query + 4 weight matrices)
//   2. bias1d[h][rel+1023] = rel_emb[bucket(rel)][h]   (131KB table)
//   3. WMMA NT-GEMMs (32x32 register tile per wave):
//        q (scaled, ->B,H,S,D), k (->B,H,S,D), v (->B,H,D,S transposed)
//   4. gate[b,h,s] from q-head dot grep_w (VALU)
//   5. fused flash attention (online softmax, bias*gate added to logits)
//   6. WMMA NT-GEMM output projection -> fp32 (S,B,E)
// ---------------------------------------------------------------------------

typedef __attribute__((ext_vector_type(16))) _Float16 v16h;
typedef __attribute__((ext_vector_type(8)))  _Float16 v8h;
typedef __attribute__((ext_vector_type(4)))  _Float16 v4h;
typedef __attribute__((ext_vector_type(8)))  float    v8f;
typedef __attribute__((ext_vector_type(4)))  float    v4f;

#define S_LEN   1024
#define BATCH   4
#define EMB     1024
#define HEADS   16
#define HDIM    64
#define NBUCK   32
#define RELTAB  2047   // 2*S-1

__device__ __forceinline__ v16h cat8(v8h lo, v8h hi) {
  return __builtin_shufflevector(lo, hi, 0,1,2,3,4,5,6,7,8,9,10,11,12,13,14,15);
}

// ---------------------------------------------------------------- converts
__global__ void f32_to_f16_kernel(const float* __restrict__ in,
                                  _Float16* __restrict__ out, int n) {
  int i = (blockIdx.x * blockDim.x + threadIdx.x) * 4;
  if (i >= n) return;
  v4f v = *(const v4f*)(in + i);
  v4h o;
  #pragma unroll
  for (int j = 0; j < 4; ++j) o[j] = (_Float16)v[j];
  *(v4h*)(out + i) = o;
}

// ------------------------------------------------- T5 relative bias table
__global__ void bias_table_kernel(const float* __restrict__ rel_emb,
                                  float* __restrict__ bias1d) {
  int idx = blockIdx.x * blockDim.x + threadIdx.x;
  if (idx >= HEADS * RELTAB) return;
  int h   = idx / RELTAB;
  int rel = idx % RELTAB - (S_LEN - 1);     // rel = k - q
  const int nb = NBUCK / 2;                 // 16
  const int max_exact = nb / 2;             // 8
  int ret = (rel > 0) ? nb : 0;
  int n = rel < 0 ? -rel : rel;
  int bucket;
  if (n < max_exact) {
    bucket = ret + n;
  } else {
    float safe = (float)(n < 1 ? 1 : n);
    // log(n/8)/log(128/8) * (16-8)
    int large = max_exact + (int)(logf(safe * 0.125f) * (8.0f / logf(16.0f)));
    if (large > nb - 1) large = nb - 1;
    bucket = ret + large;
  }
  bias1d[h * RELTAB + (rel + S_LEN - 1)] = rel_emb[bucket * HEADS + h];
}

// -------------------------------------------------------- WMMA NT-GEMM
// OUT(M x N) = A(M x K,row-major f16) * W(N x K,row-major f16)^T + bias
// mode 0: q -> (B,H,S,D) f16, *D^-0.5   mode 1: k -> (B,H,S,D) f16
// mode 2: v -> (B,H,D,S) f16 (transposed)   mode 3: -> fp32 (S,B,E)
// grid (M/64, N/64), block 128 (4 waves in 2x2); each wave owns a 32x32
// register tile (2 A-frags x 2 B-frags -> 4 WMMA accumulators), halving
// L2 bytes per WMMA vs a 16x16-per-wave layout.
__global__ void gemm_nt_kernel(const _Float16* __restrict__ A,
                               const _Float16* __restrict__ W,
                               const float* __restrict__ bias,
                               _Float16* __restrict__ outh,
                               float* __restrict__ outf, int mode) {
  const int lane = threadIdx.x & 31;
  const int wave = threadIdx.x >> 5;
  const int m0 = blockIdx.x * 64 + (wave >> 1) * 32;
  const int n0 = blockIdx.y * 64 + (wave & 1) * 32;
  const int ml = lane & 15;
  const int kh = lane >> 4;
  const _Float16* arow0 = A + (size_t)(m0 + ml) * EMB;
  const _Float16* arow1 = A + (size_t)(m0 + 16 + ml) * EMB;
  const _Float16* wrow0 = W + (size_t)(n0 + ml) * EMB;
  const _Float16* wrow1 = W + (size_t)(n0 + 16 + ml) * EMB;
  v8f acc[2][2] = {};
  for (int k = 0; k < EMB; k += 32) {
    // A-frag (16x32 f16): lane m holds 16 contiguous K values at k+kh*16
    v16h a0 = cat8(*(const v8h*)(arow0 + k + kh * 16),
                   *(const v8h*)(arow0 + k + kh * 16 + 8));
    v16h a1 = cat8(*(const v8h*)(arow1 + k + kh * 16),
                   *(const v8h*)(arow1 + k + kh * 16 + 8));
    // B-frag (32x16): lane n holds 16 contiguous K values at k+kh*16
    v16h b0 = cat8(*(const v8h*)(wrow0 + k + kh * 16),
                   *(const v8h*)(wrow0 + k + kh * 16 + 8));
    v16h b1 = cat8(*(const v8h*)(wrow1 + k + kh * 16),
                   *(const v8h*)(wrow1 + k + kh * 16 + 8));
    acc[0][0] = __builtin_amdgcn_wmma_f32_16x16x32_f16(
        false, a0, false, b0, (short)0, acc[0][0], false, false);
    acc[0][1] = __builtin_amdgcn_wmma_f32_16x16x32_f16(
        false, a0, false, b1, (short)0, acc[0][1], false, false);
    acc[1][0] = __builtin_amdgcn_wmma_f32_16x16x32_f16(
        false, a1, false, b0, (short)0, acc[1][0], false, false);
    acc[1][1] = __builtin_amdgcn_wmma_f32_16x16x32_f16(
        false, a1, false, b1, (short)0, acc[1][1], false, false);
  }
  // C layout: lane holds n = ntile + ml, rows m = mtile + r + 8*kh
  #pragma unroll
  for (int ti = 0; ti < 2; ++ti) {
    #pragma unroll
    for (int tj = 0; tj < 2; ++tj) {
      const int n = n0 + tj * 16 + ml;
      const float bn = bias[n];
      const int hh = n >> 6;       // n / HDIM
      const int d  = n & 63;       // n % HDIM
      #pragma unroll
      for (int r = 0; r < 8; ++r) {
        const int m = m0 + ti * 16 + r + 8 * kh;
        float val = acc[ti][tj][r] + bn;
        if (mode == 3) {
          outf[(size_t)m * EMB + n] = val;            // (S,B,E) fp32
        } else {
          const int s = m >> 2;                       // m / BATCH
          const int b = m & 3;                        // m % BATCH
          if (mode == 0) val *= 0.125f;               // D^-0.5
          if (mode == 2)                              // vT: (B,H,D,S)
            outh[(((size_t)b * HEADS + hh) * HDIM + d) * S_LEN + s] =
                (_Float16)val;
          else                                        // q/k: (B,H,S,D)
            outh[(((size_t)b * HEADS + hh) * S_LEN + s) * HDIM + d] =
                (_Float16)val;
        }
      }
    }
  }
}

// ------------------------------------------------------------- GRU gate
__global__ void gate_kernel(const _Float16* __restrict__ qh,
                            const float* __restrict__ grep_w,
                            const float* __restrict__ grep_b,
                            const float* __restrict__ grep_a,
                            float* __restrict__ gate) {
  int idx = blockIdx.x * blockDim.x + threadIdx.x;   // over B*H*S
  if (idx >= BATCH * HEADS * S_LEN) return;
  const int s  = idx % S_LEN;
  const int bh = idx / S_LEN;
  const int h  = bh % HEADS;
  const _Float16* qrow = qh + ((size_t)bh * S_LEN + s) * HDIM;
  float sa = 0.f, sb = 0.f;
  #pragma unroll 1
  for (int e = 0; e < 8; ++e) {
    float acc = grep_b[e];
    #pragma unroll 8
    for (int d = 0; d < HDIM; ++d) acc += (float)qrow[d] * grep_w[e * HDIM + d];
    if (e < 4) sa += acc; else sb += acc;
  }
  const float ga = 1.f / (1.f + __expf(-sa));
  const float gb = 1.f / (1.f + __expf(-sb));
  gate[idx] = ga * (gb * grep_a[h] - 1.0f) + 2.0f;
}

// -------------------------------------------- fused flash attention
// grid (B*H, S/64), block 128 (4 waves); wave owns 16 query rows.
__global__ void attn_kernel(const _Float16* __restrict__ qh,
                            const _Float16* __restrict__ kh_,
                            const _Float16* __restrict__ vT,
                            const float* __restrict__ bias1d,
                            const float* __restrict__ gate,
                            _Float16* __restrict__ ctxh) {
  __shared__ _Float16 plds[4][16 * 64];
  const int lane = threadIdx.x & 31;
  const int wave = threadIdx.x >> 5;
  const int bh = blockIdx.x;          // b*H + h
  const int h = bh & (HEADS - 1);
  const int b = bh >> 4;
  const int q0 = blockIdx.y * 64 + wave * 16;
  const int ml = lane & 15;
  const int kh2 = lane >> 4;
  const _Float16* qbase = qh  + (size_t)bh * S_LEN * HDIM;
  const _Float16* kbase = kh_ + (size_t)bh * S_LEN * HDIM;
  const _Float16* vbase = vT  + (size_t)bh * HDIM * S_LEN;
  const float* brow = bias1d + h * RELTAB;

  // preload Q A-frags covering d=[0,32) and [32,64)
  const _Float16* qrow = qbase + (size_t)(q0 + ml) * HDIM;
  v16h qa[2];
  #pragma unroll
  for (int f = 0; f < 2; ++f) {
    v8h lo = *(const v8h*)(qrow + f * 32 + kh2 * 16);
    v8h hi = *(const v8h*)(qrow + f * 32 + kh2 * 16 + 8);
    qa[f] = cat8(lo, hi);
  }
  // per-row gate values (rows r + 8*kh2)
  float grow[8];
  #pragma unroll
  for (int r = 0; r < 8; ++r)
    grow[r] = gate[(size_t)bh * S_LEN + q0 + r + 8 * kh2];

  v8f o[4] = {};                       // ctx accumulators (16q x 64d)
  float rm[8], rl[8];
  #pragma unroll
  for (int r = 0; r < 8; ++r) { rm[r] = -1e30f; rl[r] = 0.f; }

  for (int k0 = 0; k0 < S_LEN; k0 += 64) {
    // ---- S = Q K^T  (16 x 64 tile, 8 WMMAs)
    v8f sacc[4] = {};
    #pragma unroll
    for (int j = 0; j < 4; ++j) {
      const _Float16* krow = kbase + (size_t)(k0 + j * 16 + ml) * HDIM;
      #pragma unroll
      for (int f = 0; f < 2; ++f) {
        v8h b0 = *(const v8h*)(krow + f * 32 + kh2 * 16);
        v8h b1 = *(const v8h*)(krow + f * 32 + kh2 * 16 + 8);
        sacc[j] = __builtin_amdgcn_wmma_f32_16x16x32_f16(
            false, qa[f], false, cat8(b0, b1), (short)0, sacc[j], false, false);
      }
    }
    // ---- bias*gate + online softmax (fp32, half-wave shfl reductions)
    #pragma unroll
    for (int r = 0; r < 8; ++r) {
      const int q = q0 + r + 8 * kh2;
      float mx = -1e30f;
      #pragma unroll
      for (int j = 0; j < 4; ++j) {
        const int kk = k0 + j * 16 + ml;
        float v = sacc[j][r] + grow[r] * brow[kk - q + (S_LEN - 1)];
        sacc[j][r] = v;
        mx = fmaxf(mx, v);
      }
      #pragma unroll
      for (int msk = 1; msk < 16; msk <<= 1)
        mx = fmaxf(mx, __shfl_xor(mx, msk, 32));
      const float mnew = fmaxf(rm[r], mx);
      const float corr = __expf(rm[r] - mnew);
      rm[r] = mnew;
      float rs = 0.f;
      #pragma unroll
      for (int j = 0; j < 4; ++j) {
        const float p = __expf(sacc[j][r] - mnew);
        sacc[j][r] = p;
        rs += p;
      }
      #pragma unroll
      for (int msk = 1; msk < 16; msk <<= 1)
        rs += __shfl_xor(rs, msk, 32);
      rl[r] = rl[r] * corr + rs;
      #pragma unroll
      for (int j = 0; j < 4; ++j) o[j][r] *= corr;
    }
    // ---- P tile -> LDS (f16), reshaped to A-frag layout
    _Float16* pw = plds[wave];
    #pragma unroll
    for (int j = 0; j < 4; ++j)
      #pragma unroll
      for (int r = 0; r < 8; ++r)
        pw[(r + 8 * kh2) * 64 + j * 16 + ml] = (_Float16)sacc[j][r];
    __syncthreads();
    // ---- ctx += P V  (8 WMMAs; B-frags contiguous thanks to V^T layout)
    #pragma unroll
    for (int f = 0; f < 2; ++f) {
      v8h plo = *(const v8h*)(pw + ml * 64 + f * 32 + kh2 * 16);
      v8h phi = *(const v8h*)(pw + ml * 64 + f * 32 + kh2 * 16 + 8);
      v16h pa = cat8(plo, phi);
      #pragma unroll
      for (int j = 0; j < 4; ++j) {
        const _Float16* vp =
            vbase + (size_t)(j * 16 + ml) * S_LEN + k0 + f * 32 + kh2 * 16;
        v8h b0 = *(const v8h*)(vp);
        v8h b1 = *(const v8h*)(vp + 8);
        o[j] = __builtin_amdgcn_wmma_f32_16x16x32_f16(
            false, pa, false, cat8(b0, b1), (short)0, o[j], false, false);
      }
    }
    __syncthreads();
  }
  // ---- normalize, store ctx as f16 in (S,B,E) for output projection
  #pragma unroll
  for (int r = 0; r < 8; ++r) {
    const int q = q0 + r + 8 * kh2;
    const float inv = 1.0f / rl[r];
    #pragma unroll
    for (int j = 0; j < 4; ++j) {
      const int d = j * 16 + ml;
      ctxh[((size_t)q * BATCH + b) * EMB + h * HDIM + d] =
          (_Float16)(o[j][r] * inv);
    }
  }
}

// ---------------------------------------------------------------------------
extern "C" void kernel_launch(void* const* d_in, const int* in_sizes, int n_in,
                              void* d_out, int out_size, void* d_ws,
                              size_t ws_size, hipStream_t stream) {
  const float* query   = (const float*)d_in[0];
  const float* q_w     = (const float*)d_in[1];
  const float* q_b     = (const float*)d_in[2];
  const float* k_w     = (const float*)d_in[3];
  const float* k_b     = (const float*)d_in[4];
  const float* v_w     = (const float*)d_in[5];
  const float* v_b     = (const float*)d_in[6];
  const float* out_w   = (const float*)d_in[7];
  const float* out_b   = (const float*)d_in[8];
  const float* rel_emb = (const float*)d_in[9];
  const float* grep_w  = (const float*)d_in[10];
  const float* grep_b  = (const float*)d_in[11];
  const float* grep_a  = (const float*)d_in[12];

  char* ws = (char*)d_ws;
  const size_t MB = 1u << 20;
  _Float16* qhp    = (_Float16*)(ws + 0 * MB);    // (B,H,S,D)   8 MiB
  _Float16* khp    = (_Float16*)(ws + 8 * MB);    // (B,H,S,D)   8 MiB
  _Float16* vTp    = (_Float16*)(ws + 16 * MB);   // (B,H,D,S)   8 MiB
  _Float16* ctxh   = (_Float16*)(ws + 24 * MB);   // (S,B,E)     8 MiB
  _Float16* queryh = (_Float16*)(ws + 32 * MB);   // (S,B,E)     8 MiB
  _Float16* qwh    = (_Float16*)(ws + 40 * MB);   // (E,E)       2 MiB
  _Float16* kwh    = (_Float16*)(ws + 42 * MB);
  _Float16* vwh    = (_Float16*)(ws + 44 * MB);
  _Float16* owh    = (_Float16*)(ws + 46 * MB);
  float*    bias1d = (float*)   (ws + 48 * MB);   // H*2047 f32
  float*    gate   = (float*)   (ws + 49 * MB);   // B*H*S  f32

  const int nq = S_LEN * BATCH * EMB;   // 4194304
  const int nw = EMB * EMB;             // 1048576
  f32_to_f16_kernel<<<nq / 4 / 256, 256, 0, stream>>>(query, queryh, nq);
  f32_to_f16_kernel<<<nw / 4 / 256, 256, 0, stream>>>(q_w, qwh, nw);
  f32_to_f16_kernel<<<nw / 4 / 256, 256, 0, stream>>>(k_w, kwh, nw);
  f32_to_f16_kernel<<<nw / 4 / 256, 256, 0, stream>>>(v_w, vwh, nw);
  f32_to_f16_kernel<<<nw / 4 / 256, 256, 0, stream>>>(out_w, owh, nw);

  bias_table_kernel<<<(HEADS * RELTAB + 255) / 256, 256, 0, stream>>>(rel_emb,
                                                                      bias1d);
  dim3 gg(S_LEN * BATCH / 64, EMB / 64);   // (64, 16)
  gemm_nt_kernel<<<gg, 128, 0, stream>>>(queryh, qwh, q_b, qhp, nullptr, 0);
  gemm_nt_kernel<<<gg, 128, 0, stream>>>(queryh, kwh, k_b, khp, nullptr, 1);
  gemm_nt_kernel<<<gg, 128, 0, stream>>>(queryh, vwh, v_b, vTp, nullptr, 2);

  gate_kernel<<<BATCH * HEADS * S_LEN / 256, 256, 0, stream>>>(
      qhp, grep_w, grep_b, grep_a, gate);

  attn_kernel<<<dim3(BATCH * HEADS, S_LEN / 64), 128, 0, stream>>>(
      qhp, khp, vTp, bias1d, gate, ctxh);

  gemm_nt_kernel<<<gg, 128, 0, stream>>>(ctxh, owh, out_b, nullptr,
                                         (float*)d_out, 3);
}
